// AttBiMambaModel_30442728194050
// MI455X (gfx1250) — matrix operand order repaired and checked by reading, not compiled, not run on hardware
//
#include <hip/hip_runtime.h>
#include <math.h>

#define Bsz   16
#define Lseq  1024
#define Dm    512
#define DWe   300
#define DIe   1024
#define NST   64
#define Vvoc  50000
#define NTOK  (Bsz * Lseq)   // 16384

typedef __attribute__((ext_vector_type(16))) __bf16 v16bf;
typedef __attribute__((ext_vector_type(8)))  __bf16 v8bf;
typedef __attribute__((ext_vector_type(8)))  float  v8f;

enum { EPI_NONE = 0, EPI_BIAS = 1, EPI_BIAS_MASK = 2, EPI_SOFTPLUS = 3, EPI_ACC_HALF_MASK = 4 };

__device__ __forceinline__ unsigned short f2bfbits(float f) {
  unsigned int u = __builtin_bit_cast(unsigned int, f);
  u += 0x7FFFu + ((u >> 16) & 1u);           // round-to-nearest-even
  return (unsigned short)(u >> 16);
}
__device__ __forceinline__ float siluf(float x) { return x * (1.f / (1.f + __expf(-x))); }

// ---------- weight prepack: W[K][N] fp32 -> Wt[N][Kpad] bf16 (transposed, zero-padded) ----------
__global__ void pack_w_k(const float* __restrict__ W, unsigned short* __restrict__ Wt,
                         int K, int N, int Kpad) {
  int idx = blockIdx.x * blockDim.x + threadIdx.x;
  if (idx >= N * Kpad) return;
  int n = idx / Kpad, k = idx - n * Kpad;
  float v = (k < K) ? W[(size_t)k * N + n] : 0.f;
  Wt[idx] = f2bfbits(v);
}

// ---------- activation prepack: X[r][0..K) (row stride ldx) -> Y[r][Kpad] bf16, zero-padded ----------
__global__ void pack_a_k(const float* __restrict__ X, int ldx,
                         unsigned short* __restrict__ Y, int K, int Kpad, int rows) {
  int idx = blockIdx.x * blockDim.x + threadIdx.x;
  if (idx >= rows * Kpad) return;
  int r = idx / Kpad, k = idx - r * Kpad;
  float v = (k < K) ? X[(size_t)r * ldx + k] : 0.f;
  Y[idx] = f2bfbits(v);
}

__global__ void negexp_k(const float* __restrict__ a, float* __restrict__ o, int n) {
  int i = blockIdx.x * blockDim.x + threadIdx.x;
  if (i < n) o[i] = -__expf(a[i]);
}

// ---------- bf16 WMMA GEMM:  C[M,N] = epi(A[M,K] @ W[K,N]) ----------
// A bf16 [*, lda] (optionally row-gathered), W prepacked bf16 [N][Kpad].
// Wave per 16x16 tile, 2x2 tiles per 128-thread block, K stepped by 32
// (one v_wmma_f32_16x16x32_bf16 per step, fragments in contiguous k = 16*half + s order).
__global__ __launch_bounds__(128)
void gemm_bf16_wmma(const unsigned short* __restrict__ A, int lda,
                    const int* __restrict__ gather,
                    const unsigned short* __restrict__ Wt, int Kpad,
                    const float* __restrict__ bias,
                    const float* __restrict__ mask,
                    float* __restrict__ C, int ldc,
                    int M, int N, int K, int mode) {
  const int wave  = threadIdx.x >> 5;
  const int lane  = threadIdx.x & 31;
  const int mtile = (blockIdx.y * 2 + (wave >> 1)) * 16;
  const int ntile = (blockIdx.x * 2 + (wave & 1)) * 16;
  if (mtile >= M || ntile >= N) return;
  const int half = lane >> 4;     // K-half select (A and B both use k = 16*half + s)
  const int l16  = lane & 15;
  const int arow = mtile + l16;   // A: lane holds row M = lane%16
  const unsigned short* Arow = gather ? (A + (size_t)gather[arow] * lda)
                                      : (A + (size_t)arow * lda);
  const int bcol = ntile + l16;   // B: lane holds column N = lane%16
  const unsigned short* Brow = Wt + (size_t)bcol * Kpad;

  v8f acc = {};
  for (int k0 = 0; k0 < K; k0 += 32) {
    // A fragment: 16 contiguous bf16 at k0 + 16*half, loaded as two 8-element runs
    v8bf r0 = *reinterpret_cast<const v8bf*>(Arow + k0 + half * 16);
    v8bf r1 = *reinterpret_cast<const v8bf*>(Arow + k0 + half * 16 + 8);
    // B fragment: 16 contiguous bf16 (32B, 32B-aligned by construction)
    v16bf bv = *reinterpret_cast<const v16bf*>(Brow + k0 + half * 16);
    if (k0 + 32 < K) __builtin_prefetch(Brow + k0 + 32, 0, 1);   // global_prefetch_b8
    v16bf av;
#pragma unroll
    for (int e = 0; e < 8; ++e) { av[e] = r0[e]; av[8 + e] = r1[e]; }
    acc = __builtin_amdgcn_wmma_f32_16x16x32_bf16(false, av, false, bv,
                                                  (short)0, acc, false, false);
  }

  const int col = ntile + l16;
#pragma unroll
  for (int r = 0; r < 8; ++r) {
    const int row = mtile + r + half * 8;   // C/D layout: VGPR r -> M = r + 8*(lane/16)
    float v = acc[r];
    float* cp = C + (size_t)row * ldc + col;
    if (mode == EPI_NONE)           { *cp = v; }
    else if (mode == EPI_BIAS)      { *cp = v + bias[col]; }
    else if (mode == EPI_BIAS_MASK) { *cp = (v + bias[col]) * mask[row]; }
    else if (mode == EPI_SOFTPLUS)  { float t = v + bias[col];
                                      *cp = fmaxf(t, 0.f) + log1pf(__expf(-fabsf(t))); }
    else                            { *cp += 0.5f * v * mask[row]; }   // accumulate into h
  }
}

// ---------- LayerNorm over D=512, one wave per token; writes bf16 (GEMM A operand) ----------
__global__ __launch_bounds__(128)
void layernorm_k(const float* __restrict__ x, const float* __restrict__ g,
                 const float* __restrict__ b, unsigned short* __restrict__ ybf, int M) {
  int wave = threadIdx.x >> 5, lane = threadIdx.x & 31;
  int row = blockIdx.x * 4 + wave;
  if (row >= M) return;
  const float4* xr = reinterpret_cast<const float4*>(x + (size_t)row * Dm);
  uint2*       yr = reinterpret_cast<uint2*>(ybf + (size_t)row * Dm);
  const float4* gg = reinterpret_cast<const float4*>(g);
  const float4* bb = reinterpret_cast<const float4*>(b);
  float4 v[4]; float s = 0.f;
#pragma unroll
  for (int i = 0; i < 4; ++i) { v[i] = xr[lane + 32 * i]; s += v[i].x + v[i].y + v[i].z + v[i].w; }
#pragma unroll
  for (int o = 16; o > 0; o >>= 1) s += __shfl_xor(s, o, 32);
  float mu = s * (1.f / Dm);
  float q = 0.f;
#pragma unroll
  for (int i = 0; i < 4; ++i) {
    float dx = v[i].x - mu, dy = v[i].y - mu, dz = v[i].z - mu, dw = v[i].w - mu;
    q += dx * dx + dy * dy + dz * dz + dw * dw;
  }
#pragma unroll
  for (int o = 16; o > 0; o >>= 1) q += __shfl_xor(q, o, 32);
  float rs = rsqrtf(q * (1.f / Dm) + 1e-5f);
#pragma unroll
  for (int i = 0; i < 4; ++i) {
    float4 gv = gg[lane + 32 * i], bv = bb[lane + 32 * i];
    float ox = (v[i].x - mu) * rs * gv.x + bv.x;
    float oy = (v[i].y - mu) * rs * gv.y + bv.y;
    float oz = (v[i].z - mu) * rs * gv.z + bv.z;
    float ow = (v[i].w - mu) * rs * gv.w + bv.w;
    uint2 o2;
    o2.x = (unsigned int)f2bfbits(ox) | ((unsigned int)f2bfbits(oy) << 16);
    o2.y = (unsigned int)f2bfbits(oz) | ((unsigned int)f2bfbits(ow) << 16);
    yr[lane + 32 * i] = o2;
  }
}

// ---------- depthwise causal conv (k=4) + SiLU; rev flips the time window ----------
// Writes fp32 (scan/gate input) and bf16 mirror (x_proj GEMM A operand).
__global__ __launch_bounds__(256)
void conv_silu_k(const float* __restrict__ xz, const float* __restrict__ w,
                 const float* __restrict__ bias, float* __restrict__ xc,
                 unsigned short* __restrict__ xcbf, int rev) {
  int idx = blockIdx.x * 256 + threadIdx.x;
  if (idx >= NTOK * DIe) return;
  int d = idx & (DIe - 1);
  int t = (idx >> 10) & (Lseq - 1);
  int b = idx >> 20;
  const float* base = xz + (size_t)b * Lseq * (2 * DIe);
  float acc = bias[d];
#pragma unroll
  for (int k = 0; k < 4; ++k) {
    int tt = rev ? (t + 3 - k) : (t - 3 + k);
    if (tt >= 0 && tt < Lseq) acc += w[d * 4 + k] * base[(size_t)tt * (2 * DIe) + d];
  }
  float s = siluf(acc);
  xc[idx] = s;
  xcbf[idx] = f2bfbits(s);
}

// ---------- selective scan: block = (batch, 64-channel chunk); 4 lanes x 16 states per channel ----------
__global__ __launch_bounds__(256)
void scan_k(const float* __restrict__ dt, const float* __restrict__ xc,
            const float* __restrict__ xdbl, const float* __restrict__ An,
            float* __restrict__ ys, int rev) {
  __shared__ float s_dt[64], s_x[64], s_B[64], s_C[64];
  const int tid = threadIdx.x;
  const int c = tid >> 2;          // channel within chunk
  const int q = tid & 3;           // state group (16 states each)
  const int b = blockIdx.x >> 4;
  const int chunk = blockIdx.x & 15;
  const int d = chunk * 64 + c;
  float hst[16], a[16];
#pragma unroll
  for (int i = 0; i < 16; ++i) { hst[i] = 0.f; a[i] = An[(size_t)d * NST + q * 16 + i]; }
  const int role = tid >> 6;
  const int li = tid & 63;
  for (int step = 0; step < Lseq; ++step) {
    const int t = rev ? (Lseq - 1 - step) : step;
    const size_t row = (size_t)b * Lseq + t;
    float v;
    if (role == 0)      v = dt  [row * DIe + chunk * 64 + li];
    else if (role == 1) v = xc  [row * DIe + chunk * 64 + li];
    else if (role == 2) v = xdbl[row * 160 + 32 + li];   // B_t
    else                v = xdbl[row * 160 + 96 + li];   // C_t
    __syncthreads();
    if (role == 0)      s_dt[li] = v;
    else if (role == 1) s_x[li]  = v;
    else if (role == 2) s_B[li]  = v;
    else                s_C[li]  = v;
    __syncthreads();
    const float dtv = s_dt[c];
    const float dtx = dtv * s_x[c];
    float part = 0.f;
#pragma unroll
    for (int i = 0; i < 16; ++i) {
      const int n = q * 16 + i;
      hst[i] = __expf(dtv * a[i]) * hst[i] + dtx * s_B[n];
      part += hst[i] * s_C[n];
    }
    part += __shfl_xor(part, 1, 32);
    part += __shfl_xor(part, 2, 32);
    if (q == 0) ys[row * DIe + d] = part;
  }
}

// ---------- y = (ys + D*xc) * silu(z), written as bf16 (out_proj GEMM A operand) ----------
__global__ __launch_bounds__(256)
void gate_k(const float* __restrict__ ys, const float* __restrict__ xc,
            const float* __restrict__ xz, const float* __restrict__ Dp,
            unsigned short* __restrict__ ybf) {
  int idx = blockIdx.x * 256 + threadIdx.x;
  if (idx >= NTOK * DIe) return;
  int d = idx & (DIe - 1);
  int row = idx >> 10;
  float z = xz[(size_t)row * (2 * DIe) + DIe + d];
  ybf[idx] = f2bfbits((ys[idx] + Dp[d] * xc[idx]) * siluf(z));
}

// ---------- attention pooling + head ----------
__global__ void s0_k(const float* __restrict__ guide, const float* __restrict__ w1,
                     const float* __restrict__ attb, float* __restrict__ s0) {
  int lane = threadIdx.x;
  float s = 0.f;
  for (int i = lane; i < Dm; i += 32) s += guide[i] * w1[i];
#pragma unroll
  for (int o = 16; o > 0; o >>= 1) s += __shfl_xor(s, o, 32);
  if (lane == 0) s0[0] = s + attb[0];
}

__global__ __launch_bounds__(128)
void scores_k(const float* __restrict__ h, const float* __restrict__ w2,
              const float* __restrict__ mask, const float* __restrict__ s0,
              float* __restrict__ sc) {
  int wave = threadIdx.x >> 5, lane = threadIdx.x & 31;
  int row = blockIdx.x * 4 + wave;
  if (row >= NTOK) return;
  const float4* hr = reinterpret_cast<const float4*>(h + (size_t)row * Dm);
  const float4* wr = reinterpret_cast<const float4*>(w2);
  float s = 0.f;
#pragma unroll
  for (int i = 0; i < 4; ++i) {
    float4 a = hr[lane + 32 * i], b = wr[lane + 32 * i];
    s += a.x * b.x + a.y * b.y + a.z * b.z + a.w * b.w;
  }
#pragma unroll
  for (int o = 16; o > 0; o >>= 1) s += __shfl_xor(s, o, 32);
  if (lane == 0) {
    float val = s0[0] + s;
    sc[row] = (mask[row] > 0.f) ? val : -1e9f;
  }
}

__global__ __launch_bounds__(256)
void softmax_k(const float* __restrict__ sc, float* __restrict__ pr) {
  __shared__ float red[256];
  int b = blockIdx.x, tid = threadIdx.x;
  const float* s = sc + (size_t)b * Lseq;
  float mx = -3.4e38f;
  for (int i = tid; i < Lseq; i += 256) mx = fmaxf(mx, s[i]);
  red[tid] = mx; __syncthreads();
  for (int o = 128; o > 0; o >>= 1) { if (tid < o) red[tid] = fmaxf(red[tid], red[tid + o]); __syncthreads(); }
  float m = red[0]; __syncthreads();
  float sum = 0.f;
  for (int i = tid; i < Lseq; i += 256) { float e = __expf(s[i] - m); pr[(size_t)b * Lseq + i] = e; sum += e; }
  red[tid] = sum; __syncthreads();
  for (int o = 128; o > 0; o >>= 1) { if (tid < o) red[tid] += red[tid + o]; __syncthreads(); }
  float inv = 1.f / red[0];
  __syncthreads();
  for (int i = tid; i < Lseq; i += 256) pr[(size_t)b * Lseq + i] *= inv;
}

__global__ __launch_bounds__(256)
void rep_k(const float* __restrict__ h, const float* __restrict__ pr, float* __restrict__ rep) {
  __shared__ float p[Lseq];
  int b = blockIdx.x, tid = threadIdx.x;
  for (int i = tid; i < Lseq; i += 256) p[i] = pr[(size_t)b * Lseq + i];
  __syncthreads();
#pragma unroll
  for (int half = 0; half < 2; ++half) {
    int d = tid + half * 256;
    float acc = 0.f;
    for (int l = 0; l < Lseq; ++l) acc += p[l] * h[((size_t)b * Lseq + l) * Dm + d];
    rep[(size_t)b * Dm + d] = acc;
  }
}

__global__ void final_k(const float* __restrict__ rep, const float* __restrict__ pw,
                        const float* __restrict__ pb, float* __restrict__ out) {
  int b = blockIdx.x >> 1, j = blockIdx.x & 1, lane = threadIdx.x;
  float acc = 0.f;
  for (int d = lane; d < Dm; d += 32) acc += rep[(size_t)b * Dm + d] * pw[d * 2 + j];
#pragma unroll
  for (int o = 16; o > 0; o >>= 1) acc += __shfl_xor(acc, o, 32);
  if (lane == 0) out[b * 2 + j] = acc + pb[j];
}

extern "C" void kernel_launch(void* const* d_in, const int* in_sizes, int n_in,
                              void* d_out, int out_size, void* d_ws, size_t ws_size,
                              hipStream_t stream) {
  (void)out_size; (void)ws_size;
  auto F = [&](int i) { return (const float*)d_in[i]; };
  auto I = [&](int i) { return (const int*)d_in[i]; };

  // Two plausible flatten conventions: dict insertion order (in_w first, 153600)
  // or jax sorted-key order (embed_table first, 15,000,000). Detect at runtime.
  const bool sorted = (n_in > 0 && in_sizes[0] == 15000000);

  const float *embed, *masks, *attb, *guide, *w1, *w2, *in_b, *in_w, *projb, *projw;
  const int* words;
  struct MambaP { const float *in_proj, *conv_w, *conv_b, *x_proj, *dt_w, *dt_b, *A_log, *D, *out_proj; };
  MambaP mp[2][2];                    // [layer][0=fwd,1=bwd]
  const float *lng[2], *lnb[2];

  if (!sorted) {
    in_w = F(0); in_b = F(1); guide = F(2); w1 = F(3); w2 = F(4); attb = F(5);
    projw = F(6); projb = F(7);
    for (int ly = 0; ly < 2; ++ly) {
      int base = 8 + ly * 20;
      lng[ly] = F(base + 0); lnb[ly] = F(base + 1);
      for (int dir = 0; dir < 2; ++dir) {
        int mb = base + 2 + dir * 9;          // fwd first
        MambaP& m = mp[ly][dir];
        m.in_proj = F(mb + 0); m.conv_w = F(mb + 1); m.conv_b = F(mb + 2);
        m.x_proj = F(mb + 3); m.dt_w = F(mb + 4); m.dt_b = F(mb + 5);
        m.A_log = F(mb + 6); m.D = F(mb + 7); m.out_proj = F(mb + 8);
      }
    }
    embed = F(48); words = I(49); masks = F(50);
  } else {
    embed = F(0); masks = F(1); attb = F(2); guide = F(3); w1 = F(4); w2 = F(5);
    in_b = F(6); in_w = F(7);
    for (int ly = 0; ly < 2; ++ly) {
      int base = 8 + ly * 20;
      for (int dir = 0; dir < 2; ++dir) {
        int mb = base + (dir == 0 ? 9 : 0);   // bwd first in sorted order
        MambaP& m = mp[ly][dir];
        m.A_log = F(mb + 0); m.D = F(mb + 1); m.conv_b = F(mb + 2); m.conv_w = F(mb + 3);
        m.dt_b = F(mb + 4); m.dt_w = F(mb + 5); m.in_proj = F(mb + 6);
        m.out_proj = F(mb + 7); m.x_proj = F(mb + 8);
      }
      lnb[ly] = F(base + 18); lng[ly] = F(base + 19);
    }
    projb = F(48); projw = F(49); words = I(51);
  }

  // ---- workspace bump allocator (256B aligned) ----
  char* wp = (char*)d_ws;
  auto alloc = [&](size_t bytes) -> void* {
    void* p = (void*)wp;
    wp += (bytes + 255) & ~(size_t)255;
    return p;
  };
  float* h    = (float*)alloc((size_t)NTOK * Dm * 4);
  float* xzb  = (float*)alloc((size_t)NTOK * 2 * DIe * 4);
  float* xcb  = (float*)alloc((size_t)NTOK * DIe * 4);
  float* xdb  = (float*)alloc((size_t)NTOK * 160 * 4);
  float* dtb  = (float*)alloc((size_t)NTOK * DIe * 4);
  float* ysb  = (float*)alloc((size_t)NTOK * DIe * 4);
  unsigned short* embed_bf = (unsigned short*)alloc((size_t)Vvoc * 320 * 2);
  unsigned short* xln_bf   = (unsigned short*)alloc((size_t)NTOK * Dm * 2);
  unsigned short* xc_bf    = (unsigned short*)alloc((size_t)NTOK * DIe * 2);
  unsigned short* dtin_bf  = (unsigned short*)alloc((size_t)NTOK * 32 * 2);
  unsigned short* ys_bf    = (unsigned short*)alloc((size_t)NTOK * DIe * 2);
  unsigned short* pw_inw   = (unsigned short*)alloc((size_t)512 * 320 * 2);
  float* An[2][2];
  unsigned short *pw_ip[2][2], *pw_xp[2][2], *pw_dw[2][2], *pw_op[2][2];
  for (int ly = 0; ly < 2; ++ly)
    for (int dir = 0; dir < 2; ++dir) {
      An[ly][dir]    = (float*)alloc((size_t)DIe * NST * 4);
      pw_ip[ly][dir] = (unsigned short*)alloc((size_t)2048 * 512 * 2);
      pw_xp[ly][dir] = (unsigned short*)alloc((size_t)160 * 1024 * 2);
      pw_dw[ly][dir] = (unsigned short*)alloc((size_t)1024 * 32 * 2);
      pw_op[ly][dir] = (unsigned short*)alloc((size_t)512 * 1024 * 2);
    }
  float* s0v  = (float*)alloc(256);
  float* scr  = (float*)alloc((size_t)NTOK * 4);
  float* prb  = (float*)alloc((size_t)NTOK * 4);
  float* repb = (float*)alloc((size_t)Bsz * Dm * 4);

  // ---- prepack (deterministic, every call) ----
  pack_a_k<<<((size_t)Vvoc * 320 + 255) / 256, 256, 0, stream>>>(embed, DWe, embed_bf, DWe, 320, Vvoc);
  pack_w_k<<<(512 * 320 + 255) / 256, 256, 0, stream>>>(in_w, pw_inw, 300, 512, 320);
  for (int ly = 0; ly < 2; ++ly)
    for (int dir = 0; dir < 2; ++dir) {
      const MambaP& m = mp[ly][dir];
      pack_w_k<<<(2048 * 512 + 255) / 256, 256, 0, stream>>>(m.in_proj, pw_ip[ly][dir], 512, 2048, 512);
      pack_w_k<<<(160 * 1024 + 255) / 256, 256, 0, stream>>>(m.x_proj, pw_xp[ly][dir], 1024, 160, 1024);
      pack_w_k<<<(1024 * 32 + 255) / 256, 256, 0, stream>>>(m.dt_w, pw_dw[ly][dir], 32, 1024, 32);
      pack_w_k<<<(512 * 1024 + 255) / 256, 256, 0, stream>>>(m.out_proj, pw_op[ly][dir], 1024, 512, 1024);
      negexp_k<<<(DIe * NST + 255) / 256, 256, 0, stream>>>(m.A_log, An[ly][dir], DIe * NST);
    }

  // ---- embedding gather + input projection (WMMA, gathered bf16 A rows, K padded to 320) ----
  gemm_bf16_wmma<<<dim3(Dm / 32, NTOK / 32), 128, 0, stream>>>(
      embed_bf, 320, words, pw_inw, 320, in_b, masks, h, Dm, NTOK, Dm, 320, EPI_BIAS_MASK);

  for (int ly = 0; ly < 2; ++ly) {
    layernorm_k<<<NTOK / 4, 128, 0, stream>>>(h, lng[ly], lnb[ly], xln_bf, NTOK);
    for (int dir = 0; dir < 2; ++dir) {
      const MambaP& m = mp[ly][dir];
      const int rev = dir;   // bwd = reversed conv window + reversed scan (flips cancel)
      gemm_bf16_wmma<<<dim3(2 * DIe / 32, NTOK / 32), 128, 0, stream>>>(
          xln_bf, Dm, nullptr, pw_ip[ly][dir], 512, nullptr, nullptr,
          xzb, 2 * DIe, NTOK, 2 * DIe, Dm, EPI_NONE);
      conv_silu_k<<<(NTOK * DIe) / 256, 256, 0, stream>>>(xzb, m.conv_w, m.conv_b, xcb, xc_bf, rev);
      gemm_bf16_wmma<<<dim3(160 / 32, NTOK / 32), 128, 0, stream>>>(
          xc_bf, DIe, nullptr, pw_xp[ly][dir], 1024, nullptr, nullptr,
          xdb, 160, NTOK, 160, DIe, EPI_NONE);
      pack_a_k<<<(NTOK * 32 + 255) / 256, 256, 0, stream>>>(xdb, 160, dtin_bf, 32, 32, NTOK);
      gemm_bf16_wmma<<<dim3(DIe / 32, NTOK / 32), 128, 0, stream>>>(
          dtin_bf, 32, nullptr, pw_dw[ly][dir], 32, m.dt_b, nullptr,
          dtb, DIe, NTOK, DIe, 32, EPI_SOFTPLUS);
      scan_k<<<Bsz * 16, 256, 0, stream>>>(dtb, xcb, xdb, An[ly][dir], ysb, rev);
      gate_k<<<(NTOK * DIe) / 256, 256, 0, stream>>>(ysb, xcb, xzb, m.D, ys_bf);
      gemm_bf16_wmma<<<dim3(Dm / 32, NTOK / 32), 128, 0, stream>>>(
          ys_bf, DIe, nullptr, pw_op[ly][dir], 1024, nullptr, masks,
          h, Dm, NTOK, Dm, DIe, EPI_ACC_HALF_MASK);
    }
  }

  s0_k<<<1, 32, 0, stream>>>(guide, w1, attb, s0v);
  scores_k<<<NTOK / 4, 128, 0, stream>>>(h, w2, masks, s0v, scr);
  softmax_k<<<Bsz, 256, 0, stream>>>(scr, prb);
  rep_k<<<Bsz, 256, 0, stream>>>(h, prb, repb);
  final_k<<<Bsz * 2, 32, 0, stream>>>(repb, projw, projb, (float*)d_out);
}
